// NodeAttention_40613210751284
// MI455X (gfx1250) — matrix operand order repaired and checked by reading, not compiled, not run on hardware
//
#include <hip/hip_runtime.h>
#include <hip/hip_bf16.h>

typedef __attribute__((ext_vector_type(16))) __bf16 bf16x16;
typedef __attribute__((ext_vector_type(8)))  float  f32x8;
typedef __attribute__((ext_vector_type(4)))  int    v4i;

#if defined(__has_builtin)
#  if __has_builtin(__builtin_amdgcn_global_load_async_to_lds_b128) && \
      __has_builtin(__builtin_amdgcn_s_wait_asynccnt)
#    define HAVE_ASYNC_LDS 1
#  endif
#endif
#ifndef HAVE_ASYNC_LDS
#  define HAVE_ASYNC_LDS 0
#endif

namespace {

constexpr int NNODE = 16384;
constexpr int EEXP  = 4096;
constexpr int DDIM  = 512;
constexpr int TP    = 40;   // LDS tile pitch (bf16 elems) = 80B: 16B-aligned, bank-skewed

enum { SRC_F32 = 0, SRC_BF16 = 1 };
enum { OUT_BF16 = 0, OUT_BF16_T = 1, OUT_F32_SCALE = 2, OUT_F32_ELU = 3 };

__device__ __forceinline__ unsigned short f2bf_rne(float f) {
  unsigned u = __float_as_uint(f);
  u += 0x7FFFu + ((u >> 16) & 1u);
  return (unsigned short)(u >> 16);
}

union FragBF { uint4 q[2]; bf16x16 v; };
union U4S8  { uint4 q; unsigned short s[8]; };

__device__ __forceinline__ uint4 cvt8_bf16(float4 f0, float4 f1) {
  U4S8 u;
  u.s[0]=f2bf_rne(f0.x); u.s[1]=f2bf_rne(f0.y); u.s[2]=f2bf_rne(f0.z); u.s[3]=f2bf_rne(f0.w);
  u.s[4]=f2bf_rne(f1.x); u.s[5]=f2bf_rne(f1.y); u.s[6]=f2bf_rne(f1.z); u.s[7]=f2bf_rne(f1.w);
  return u.q;
}

// Stage one 128x32 tile side into LDS. Thread handles rows sr and sr+64, k-cols [sc, sc+8).
// bf16 source + HAVE_ASYNC_LDS -> GLOBAL_LOAD_ASYNC_TO_LDS_B128 (ASYNCcnt),
// f32 source -> load + RNE-convert + ds_store.
template<int SRC>
__device__ __forceinline__ void stage_issue(const void* __restrict__ G, int ld, int base_row,
                                            int k0, int K, unsigned short* S, int sr, int sc)
{
#pragma unroll
  for (int hrow = 0; hrow < 2; ++hrow) {
    const int row = sr + hrow * 64;
    if (SRC == SRC_F32) {
      const float* gp = (const float*)G + (size_t)(base_row + row) * ld + (k0 + sc);
      float4 f0 = *(const float4*)gp;
      float4 f1 = *(const float4*)(gp + 4);
      *(uint4*)&S[row * TP + sc] = cvt8_bf16(f0, f1);
      if (k0 + 32 < K) __builtin_prefetch(gp + 32, 0, 1);
    } else {
      const unsigned short* gp =
          (const unsigned short*)G + (size_t)(base_row + row) * ld + (k0 + sc);
#if HAVE_ASYNC_LDS
      v4i* gq = (v4i*)gp;                       // drop const / retype (generic)
      v4i* lq = (v4i*)&S[row * TP + sc];
      __builtin_amdgcn_global_load_async_to_lds_b128(
          (__attribute__((address_space(1))) v4i*)gq,
          (__attribute__((address_space(3))) v4i*)lq, 0, 0);
#else
      *(uint4*)&S[row * TP + sc] = *(const uint4*)gp;
      if (k0 + 32 < K) __builtin_prefetch(gp + 32, 0, 1);
#endif
    }
  }
}

__device__ __forceinline__ f32x8 wmma_bf16(const FragBF& a, const FragBF& b, f32x8 c) {
  return __builtin_amdgcn_wmma_f32_16x16x32_bf16(false, a.v, false, b.v,
                                                 (short)0, c, false, false);
}

// Per-wave 32x64 compute on one K-chunk: 2 A-fragments, 4 B-fragments, 8 WMMAs.
// Fragment layouts per ISA 7.12.2 (16-bit A: lanes0-15 K{0-7,16-23}, lanes16-31 K{8-15,24-31};
// 16-bit B: same K order as A).
__device__ __forceinline__ void wave_compute(const unsigned short* As, const unsigned short* Bs,
                                             int mi, int nb, int lane, f32x8* acc)
{
  const int m = lane & 15, h = lane >> 4;
  FragBF fa0, fa1;
  {
    const unsigned short* p0 = &As[(mi + m) * TP + h * 8];
    fa0.q[0] = *(const uint4*)p0;
    fa0.q[1] = *(const uint4*)(p0 + 16);
    const unsigned short* p1 = &As[(mi + 16 + m) * TP + h * 8];
    fa1.q[0] = *(const uint4*)p1;
    fa1.q[1] = *(const uint4*)(p1 + 16);
  }
#pragma unroll
  for (int jn = 0; jn < 4; ++jn) {
    FragBF fb;
    const unsigned short* pb = &Bs[(nb + jn * 16 + m) * TP + h * 8];
    fb.q[0] = *(const uint4*)pb;
    fb.q[1] = *(const uint4*)(pb + 16);
    acc[jn * 2 + 0] = wmma_bf16(fa0, fb, acc[jn * 2 + 0]);
    acc[jn * 2 + 1] = wmma_bf16(fa1, fb, acc[jn * 2 + 1]);
  }
}

// ---------------------------------------------------------------------------
// NT GEMM: C[m,n] = post( scale * sum_k A[m,k] * B[n,k] )
// A: M x K row-major, B: Nc x K row-major. 128x128 block tile, 8 waves,
// 32x64 per-wave tile. Double-buffered async global->LDS staging when available.
// ---------------------------------------------------------------------------
template<int ASRC, int BSRC, int OMODE>
__global__ __launch_bounds__(256)
void gemm_nt_wmma(const void* __restrict__ Ag, const void* __restrict__ Bg,
                  void* __restrict__ Cg, void* __restrict__ CgT,
                  int M, int Nc, int K, float scale)
{
#if HAVE_ASYNC_LDS
  __shared__ __align__(16) unsigned short As[2][128 * TP];
  __shared__ __align__(16) unsigned short Bs[2][128 * TP];
#else
  __shared__ __align__(16) unsigned short As[1][128 * TP];
  __shared__ __align__(16) unsigned short Bs[1][128 * TP];
#endif

  const int t    = threadIdx.x;
  const int lane = t & 31;
  const int w    = t >> 5;
  const int mi   = (w & 3) * 32;     // wave 32-row band in 128-row block tile
  const int nb   = (w >> 2) * 64;    // wave 64-col band in 128-col block tile
  const int row0 = blockIdx.y * 128;
  const int col0 = blockIdx.x * 128;

  const int sr = t >> 2;             // staging row 0..63 (and +64)
  const int sc = (t & 3) * 8;        // staging k-col {0,8,16,24}

  f32x8 acc[8];
#pragma unroll
  for (int i = 0; i < 8; ++i) acc[i] = f32x8{0.f,0.f,0.f,0.f,0.f,0.f,0.f,0.f};

  const int nk = K / 32;

#if HAVE_ASYNC_LDS
  // async ops issued per stage_issue pair (A side + B side) per thread
  constexpr int NASYNC = (ASRC == SRC_BF16 ? 2 : 0) + (BSRC == SRC_BF16 ? 2 : 0);
  stage_issue<ASRC>(Ag, K, row0, 0, K, As[0], sr, sc);
  stage_issue<BSRC>(Bg, K, col0, 0, K, Bs[0], sr, sc);
  for (int ck = 0; ck < nk; ++ck) {
    const int cur = ck & 1;
    if (ck + 1 < nk) {
      stage_issue<ASRC>(Ag, K, row0, (ck + 1) * 32, K, As[1 - cur], sr, sc);
      stage_issue<BSRC>(Bg, K, col0, (ck + 1) * 32, K, Bs[1 - cur], sr, sc);
      __builtin_amdgcn_s_wait_asynccnt(NASYNC);   // chunk ck complete (in-order)
    } else {
      __builtin_amdgcn_s_wait_asynccnt(0);
    }
    __syncthreads();                 // all waves' data for chunk ck visible
    wave_compute(As[cur], Bs[cur], mi, nb, lane, acc);
    __syncthreads();                 // readers done before buffer reuse
  }
#else
  for (int ck = 0; ck < nk; ++ck) {
    __syncthreads();                 // previous chunk's readers done
    stage_issue<ASRC>(Ag, K, row0, ck * 32, K, As[0], sr, sc);
    stage_issue<BSRC>(Bg, K, col0, ck * 32, K, Bs[0], sr, sc);
    __syncthreads();                 // staged data visible
    wave_compute(As[0], Bs[0], mi, nb, lane, acc);
  }
#endif

  // ---- epilogue: C layout = VGPR g, lanes 0-15 -> M=sub g, lanes 16-31 -> M=sub 8+g
  const int n  = lane & 15;
  const int mh = (lane >> 4) * 8;
#pragma unroll
  for (int jn = 0; jn < 4; ++jn) {
#pragma unroll
    for (int ai = 0; ai < 2; ++ai) {
      const f32x8 a = acc[jn * 2 + ai];
#pragma unroll
      for (int g = 0; g < 8; ++g) {
        const size_t m    = (size_t)(row0 + mi + ai * 16 + mh + g);
        const size_t cidx = (size_t)(col0 + nb + jn * 16 + n);
        const float  v    = a[g] * scale;
        if (OMODE == OUT_F32_SCALE) {
          ((float*)Cg)[m * Nc + cidx] = v;
        } else if (OMODE == OUT_F32_ELU) {
          ((float*)Cg)[m * Nc + cidx] = (v > 0.f) ? v : (__expf(v) - 1.0f);
        } else {
          const unsigned short b = f2bf_rne(v);
          ((unsigned short*)Cg)[m * Nc + cidx] = b;
          if (OMODE == OUT_BF16_T)                  // also write transposed copy
            ((unsigned short*)CgT)[cidx * (size_t)M + m] = b;
        }
      }
    }
  }
}

// ---------------------------------------------------------------------------
// Row-wise masked softmax over E, in place on the B output region.
// B = softmax(where(H>0, logits, -inf)) * (H>0); exact zeros for masked slots.
// ---------------------------------------------------------------------------
__global__ __launch_bounds__(256)
void masked_softmax(float* __restrict__ B, const int* __restrict__ H)
{
  __shared__ float sd[256];
  const int row = blockIdx.x;
  const int t   = threadIdx.x;
  const size_t base = (size_t)row * EEXP;

  float lv[16];
  unsigned mb = 0u;
  float mx = -3.0e38f;
#pragma unroll
  for (int i = 0; i < 16; ++i) {
    const int idx = t + i * 256;
    float l = B[base + idx];
    const int h = H[base + idx];
    if (h > 0) mb |= (1u << i); else l = -3.0e38f;
    lv[i] = l;
    mx = fmaxf(mx, l);
  }
  sd[t] = mx; __syncthreads();
  for (int s = 128; s > 0; s >>= 1) {
    if (t < s) sd[t] = fmaxf(sd[t], sd[t + s]);
    __syncthreads();
  }
  mx = sd[0];
  __syncthreads();

  float sum = 0.f;
#pragma unroll
  for (int i = 0; i < 16; ++i) {
    const float e = __expf(lv[i] - mx);
    lv[i] = e;
    sum += e;
  }
  sd[t] = sum; __syncthreads();
  for (int s = 128; s > 0; s >>= 1) {
    if (t < s) sd[t] += sd[t + s];
    __syncthreads();
  }
  const float inv = (sd[0] > 0.f) ? (1.0f / sd[0]) : 0.f;

#pragma unroll
  for (int i = 0; i < 16; ++i) {
    const float p = ((mb >> i) & 1u) ? lv[i] * inv : 0.f;
    B[base + t + i * 256] = p;
  }
}

} // namespace

// ---------------------------------------------------------------------------
// d_in: [0]=x (N,D) f32, [1]=e_repr (E,D) f32, [2]=H (N,E) i32,
//       [3]=W_e (D,D) f32, [4]=W_n (D,D) f32
// d_out: z_h (N*D f32) then B (N*E f32), flat-concatenated.
// d_ws: hn_bf (N*D) + he_bf (E*D) + heT_bf (D*E), bf16 -> 24 MB total.
// ---------------------------------------------------------------------------
extern "C" void kernel_launch(void* const* d_in, const int* in_sizes, int n_in,
                              void* d_out, int out_size, void* d_ws, size_t ws_size,
                              hipStream_t stream)
{
  (void)in_sizes; (void)n_in; (void)out_size; (void)ws_size;
  const float* x  = (const float*)d_in[0];
  const float* er = (const float*)d_in[1];
  const int*   H  = (const int*)d_in[2];
  const float* We = (const float*)d_in[3];
  const float* Wn = (const float*)d_in[4];

  float* z  = (float*)d_out;
  float* Bp = (float*)d_out + (size_t)NNODE * DDIM;

  unsigned short* hn_bf  = (unsigned short*)d_ws;               // N x D
  unsigned short* he_bf  = hn_bf + (size_t)NNODE * DDIM;        // E x D
  unsigned short* heT_bf = he_bf + (size_t)EEXP * DDIM;         // D x E (transposed)

  const float inv_sqrt_d = 0.044194173824159216f;   // 1/sqrt(512)

  // he = e_repr @ W_e^T  -> bf16 (E x D) + transposed copy (D x E)
  gemm_nt_wmma<SRC_F32, SRC_F32, OUT_BF16_T>
      <<<dim3(DDIM/128, EEXP/128), 256, 0, stream>>>(
      er, We, he_bf, heT_bf, EEXP, DDIM, DDIM, 1.0f);
  // hn = x @ W_n^T       -> bf16 (N x D)
  gemm_nt_wmma<SRC_F32, SRC_F32, OUT_BF16>
      <<<dim3(DDIM/128, NNODE/128), 256, 0, stream>>>(
      x, Wn, hn_bf, nullptr, NNODE, DDIM, DDIM, 1.0f);
  // logits = (hn @ he^T) / sqrt(D)  -> f32 into B output region
  gemm_nt_wmma<SRC_BF16, SRC_BF16, OUT_F32_SCALE>
      <<<dim3(EEXP/128, NNODE/128), 256, 0, stream>>>(
      hn_bf, he_bf, Bp, nullptr, NNODE, EEXP, DDIM, inv_sqrt_d);
  // masked softmax rows (in place)
  masked_softmax<<<NNODE, 256, 0, stream>>>(Bp, H);
  // z = elu(B @ he) == NT GEMM against heT (D x E), ELU fused
  gemm_nt_wmma<SRC_F32, SRC_BF16, OUT_F32_ELU>
      <<<dim3(DDIM/128, NNODE/128), 256, 0, stream>>>(
      Bp, heT_bf, z, nullptr, NNODE, DDIM, EEXP, 1.0f);
}
